// DifferentiablePianoSynth_532575944791
// MI455X (gfx1250) — matrix operand order repaired and checked by reading, not compiled, not run on hardware
//
#include <hip/hip_runtime.h>
#include <hip/hip_bf16.h>
#include <math.h>

typedef __attribute__((ext_vector_type(16))) _Float16 v16h;
typedef __attribute__((ext_vector_type(8)))  _Float16 v8h;
typedef __attribute__((ext_vector_type(8)))  float    v8f;
typedef __attribute__((ext_vector_type(4)))  float    v4f;

#define BATCH 8192
#define NNOTE 88
#define NPART 8
#define NP    704      // NNOTE*NPART
#define TT    96
#define SRATE 48000.0f
#define TWO_PI_F 6.283185307179586f

// ---------------------------------------------------------------------------
// helpers
// ---------------------------------------------------------------------------
__device__ __forceinline__ float sigf(float x) { return 1.0f / (1.0f + __expf(-x)); }

__device__ __forceinline__ float rem2pi(float x) {
    // jnp.remainder(x, 2pi): result in [0, 2pi)
    return x - floorf(x * (1.0f / TWO_PI_F)) * TWO_PI_F;
}

// edge weight of note m from summing the 'same'-padded 5-tap conv over notes
__device__ __forceinline__ float note_weight(int m) {
    const float k[5] = {0.05f, 0.15f, 0.35f, 0.3f, 0.15f};
    float w = 0.0f;
#pragma unroll
    for (int j = 0; j < 5; ++j) {
        int n = m - j + 2;
        if (n >= 0 && n < NNOTE) w += k[j];
    }
    return w;
}

__device__ __forceinline__ unsigned int mix32(unsigned int x) {
    x ^= x >> 17; x *= 0xED5AD4BBu;
    x ^= x >> 11; x *= 0xAC4C1B51u;
    x ^= x >> 15; x *= 0x31848BABu;
    x ^= x >> 14;
    return x;
}

// counter-based standard normal for (t, b, n)
__device__ __forceinline__ float rand_normal(int t, int b, int n) {
    unsigned int h1 = mix32(0x9E3779B9u * (unsigned)t + 0x85EBCA6Bu * (unsigned)b +
                            0xC2B2AE35u * (unsigned)n + 42u);
    unsigned int h2 = mix32(h1 ^ 0x68BC21EBu);
    float u1 = (float)(h1 >> 8) * 5.9604645e-08f + 2.9802322e-08f; // (0,1)
    float u2 = (float)(h2 >> 8) * 5.9604645e-08f;
    float r  = sqrtf(-2.0f * __logf(u1));
    return r * __cosf(TWO_PI_F * u2);
}

// ---------------------------------------------------------------------------
// K1: per-(b,n) prep. Writes gate/onset/amp_f/noise_f/phase_f outputs and the
// f16 A1/A2 GEMM operands (row-major [b][704]) + weighted noise state.
// ---------------------------------------------------------------------------
__global__ __launch_bounds__(256)
void piano_prep_kernel(const float* __restrict__ note_logits,
                       const float* __restrict__ velocity,
                       const float* __restrict__ brightness,
                       const float* __restrict__ noise_amt,
                       const float* __restrict__ amp_state,
                       const float* __restrict__ phase_state,
                       const float* __restrict__ noise_state,
                       const float* __restrict__ prev_gate,
                       const float* __restrict__ harmonic_logits,
                       const float* __restrict__ decay_logits,
                       const float* __restrict__ noise_decay_logits,
                       _Float16* __restrict__ A1, _Float16* __restrict__ A2,
                       float* __restrict__ nw,
                       float* __restrict__ out_ampf, float* __restrict__ out_phasef,
                       float* __restrict__ out_noisef, float* __restrict__ out_gate,
                       float* __restrict__ out_onset) {
    int idx = blockIdx.x * blockDim.x + threadIdx.x;
    if (idx >= BATCH * NNOTE) return;
    int n = idx % NNOTE;

    float g  = sigf(note_logits[idx]);
    float on = fmaxf(g - prev_gate[idx], 0.0f);
    out_gate[idx]  = g;
    out_onset[idx] = on;

    float nd  = 0.9995f + 0.00049f * sigf(decay_logits[n]);
    float nsd = 0.9f + 0.09f * sigf(noise_decay_logits[n]);

    float amp0   = fmaxf(amp_state[idx] * nd, g * velocity[idx]);
    float noise0 = noise_state[idx] * nsd + on * noise_amt[idx];

    out_ampf[idx]   = amp0 * __powf(nd, (float)TT);
    out_noisef[idx] = noise0 * __powf(nsd, (float)TT);
    nw[idx] = note_weight(n) * noise0;

    // harmonic weights: softmax over P, tilt by brightness, renormalize
    float br = brightness[idx];
    float hl[NPART], hmax = -1e30f;
#pragma unroll
    for (int p = 0; p < NPART; ++p) { hl[p] = harmonic_logits[n * NPART + p]; hmax = fmaxf(hmax, hl[p]); }
    float se = 0.0f, e[NPART];
#pragma unroll
    for (int p = 0; p < NPART; ++p) { e[p] = __expf(hl[p] - hmax); se += e[p]; }
    float inv_se = 1.0f / se;
    float slope = 1.5f - 1.2f * br;
    float w[NPART], sw = 0.0f;
#pragma unroll
    for (int p = 0; p < NPART; ++p) {
        w[p] = e[p] * inv_se * __expf(-(float)(p + 1) * slope);
        sw += w[p];
    }
    float inv_sw = 1.0f / (sw + 1e-8f);

    float midi = 21.0f + (float)n;
    float freq = 440.0f * __powf(2.0f, (midi - 69.0f) * (1.0f / 12.0f));

    int row = idx / NNOTE;                 // batch index
    v8h a1v, a2v;
    v4f phlo, phhi;
#pragma unroll
    for (int p = 0; p < NPART; ++p) {
        float theta0 = phase_state[idx * NPART + p];
        float dinc   = TWO_PI_F * freq * (float)(p + 1) / SRATE;
        float phf    = rem2pi(theta0 + (float)TT * dinc);
        if (p < 4) phlo[p] = phf; else phhi[p - 4] = phf;
        float s0, c0;
        __sincosf(theta0, &s0, &c0);
        float ah = amp0 * w[p] * inv_sw;
        a1v[p] = (_Float16)(ah * s0);
        a2v[p] = (_Float16)(ah * c0);
    }
    // packed 16B stores (np = n*8+p contiguous per thread)
    *(v8h*)(A1 + (size_t)row * NP + n * NPART) = a1v;
    *(v8h*)(A2 + (size_t)row * NP + n * NPART) = a2v;
    *(v4f*)(out_phasef + (size_t)idx * NPART)     = phlo;
    *(v4f*)(out_phasef + (size_t)idx * NPART + 4) = phhi;
}

// ---------------------------------------------------------------------------
// K2: batch-independent basis C1t/C2t, transposed [t][704] for B-fragments,
// plus the noise-decay power table ndpow[t][n].
// ---------------------------------------------------------------------------
__global__ __launch_bounds__(256)
void piano_basis_kernel(const float* __restrict__ decay_logits,
                        const float* __restrict__ noise_decay_logits,
                        _Float16* __restrict__ C1t, _Float16* __restrict__ C2t,
                        float* __restrict__ ndpow) {
    int idx = blockIdx.x * blockDim.x + threadIdx.x;
    if (idx >= NP * TT) return;
    int np = idx % NP;
    int t  = idx / NP;
    int n  = np / NPART;
    int p  = np % NPART;

    float nd   = 0.9995f + 0.00049f * sigf(decay_logits[n]);
    float midi = 21.0f + (float)n;
    float freq = 440.0f * __powf(2.0f, (midi - 69.0f) * (1.0f / 12.0f));
    float dinc = TWO_PI_F * freq * (float)(p + 1) / SRATE;

    float dpt = __powf(nd, (float)t);
    float s, c;
    __sincosf((float)(t + 1) * dinc, &s, &c);
    C1t[t * NP + np] = (_Float16)(dpt * c);
    C2t[t * NP + np] = (_Float16)(dpt * s);

    if (p == 0) {
        float nsd = 0.9f + 0.09f * sigf(noise_decay_logits[n]);
        ndpow[t * NNOTE + n] = __powf(nsd, (float)t);
    }
}

// ---------------------------------------------------------------------------
// K3: noise path. For each (b,t): noise_gain * sum_n nw[b,n]*nsd^t*eps(t,b,n).
// Writes the noise term into audio; the WMMA kernel adds tonal on top.
// ---------------------------------------------------------------------------
__global__ __launch_bounds__(256)
void piano_noise_kernel(const float* __restrict__ nw,
                        const float* __restrict__ ndpow,
                        const float* __restrict__ noise_gain,
                        float* __restrict__ audio) {
    int b = blockIdx.x * blockDim.x + threadIdx.x;
    int t = blockIdx.y;
    if (b >= BATCH) return;
    float acc = 0.0f;
#pragma unroll 4
    for (int n = 0; n < NNOTE; ++n) {
        acc = fmaf(nw[b * NNOTE + n] * ndpow[t * NNOTE + n], rand_normal(t, b, n), acc);
    }
    audio[b * TT + t] = noise_gain[0] * acc;
}

// ---------------------------------------------------------------------------
// K4: tonal GEMM via v_wmma_f32_16x16x32_f16.
// One wave per 16(b) x 32(t) tile pair: A fragments loaded once per k-chunk,
// reused across both t-tiles (halves the dominant A traffic vs 16x16 tiles).
// K = 704 in 22 chunks of 32; 4 WMMAs per chunk.
// ---------------------------------------------------------------------------
__device__ __forceinline__ v16h load_a_frag(const _Float16* __restrict__ A,
                                            int rowBase, int lane, int k0) {
    int m = lane & 15;
    int off = (lane < 16) ? 0 : 8;
    const _Float16* p = A + (size_t)(rowBase + m) * NP + k0 + off;
    v8h lo = *(const v8h*)(p);
    v8h hi = *(const v8h*)(p + 16);
    return __builtin_shufflevector(lo, hi, 0, 1, 2, 3, 4, 5, 6, 7,
                                          8, 9, 10, 11, 12, 13, 14, 15);
}

__device__ __forceinline__ v16h load_b_frag(const _Float16* __restrict__ Ct,
                                            int colBase, int lane, int k0) {
    int nn = lane & 15;
    int off = (lane < 16) ? 0 : 8;
    const _Float16* p = Ct + (size_t)(colBase + nn) * NP + k0 + off;
    v8h lo = *(const v8h*)(p);
    v8h hi = *(const v8h*)(p + 16);
    return __builtin_shufflevector(lo, hi, 0, 1, 2, 3, 4, 5, 6, 7,
                                          8, 9, 10, 11, 12, 13, 14, 15);
}

__global__ __launch_bounds__(32)
void piano_wmma_kernel(const _Float16* __restrict__ A1, const _Float16* __restrict__ A2,
                       const _Float16* __restrict__ C1t, const _Float16* __restrict__ C2t,
                       const float* __restrict__ master_gain,
                       float* __restrict__ audio) {
    const int tPairs = TT / 32;                       // 3
    int tile  = blockIdx.x;                           // 0 .. 512*3-1
    int tileB = tile / tPairs;
    int tPair = tile % tPairs;
    int lane  = threadIdx.x;

    int rowBase = tileB * 16;
    int colBase = tPair * 32;

    v8f acc0 = {};
    v8f acc1 = {};
#pragma unroll 2
    for (int k0 = 0; k0 < NP; k0 += 32) {
        v16h a1 = load_a_frag(A1, rowBase, lane, k0);
        v16h a2 = load_a_frag(A2, rowBase, lane, k0);
        v16h b10 = load_b_frag(C1t, colBase,      lane, k0);
        v16h b11 = load_b_frag(C1t, colBase + 16, lane, k0);
        v16h b20 = load_b_frag(C2t, colBase,      lane, k0);
        v16h b21 = load_b_frag(C2t, colBase + 16, lane, k0);
        acc0 = __builtin_amdgcn_wmma_f32_16x16x32_f16(false, a1, false, b10,
                                                      (short)0, acc0, false, false);
        acc1 = __builtin_amdgcn_wmma_f32_16x16x32_f16(false, a1, false, b11,
                                                      (short)0, acc1, false, false);
        acc0 = __builtin_amdgcn_wmma_f32_16x16x32_f16(false, a2, false, b20,
                                                      (short)0, acc0, false, false);
        acc1 = __builtin_amdgcn_wmma_f32_16x16x32_f16(false, a2, false, b21,
                                                      (short)0, acc1, false, false);
    }

    float mg = master_gain[0];
    int nn = lane & 15;
    int mOff = (lane < 16) ? 0 : 8;
#pragma unroll
    for (int r = 0; r < 8; ++r) {
        int b = rowBase + mOff + r;
        float* dst0 = audio + (size_t)b * TT + colBase + nn;
        float* dst1 = dst0 + 16;
        *dst0 = fmaf(mg, acc0[r], *dst0);   // noise term already there (K3)
        *dst1 = fmaf(mg, acc1[r], *dst1);
    }
}

// ---------------------------------------------------------------------------
// launch
// ---------------------------------------------------------------------------
extern "C" void kernel_launch(void* const* d_in, const int* in_sizes, int n_in,
                              void* d_out, int out_size, void* d_ws, size_t ws_size,
                              hipStream_t stream) {
    (void)in_sizes; (void)n_in; (void)out_size; (void)ws_size;

    const float* note_logits   = (const float*)d_in[0];
    const float* velocity      = (const float*)d_in[1];
    const float* brightness    = (const float*)d_in[2];
    const float* noise_amt     = (const float*)d_in[3];
    const float* amp_state     = (const float*)d_in[4];
    const float* phase_state   = (const float*)d_in[5];
    const float* noise_state   = (const float*)d_in[6];
    const float* prev_gate     = (const float*)d_in[7];
    const float* harmonic_log  = (const float*)d_in[8];
    const float* decay_logits  = (const float*)d_in[9];
    const float* noise_dec_log = (const float*)d_in[10];
    const float* master_gain   = (const float*)d_in[11];
    const float* noise_gain    = (const float*)d_in[12];

    float* out = (float*)d_out;
    float* audio    = out;                                        // [B,1,T]
    float* out_ampf = audio + BATCH * TT;                         // [B,N]
    float* out_phf  = out_ampf + BATCH * NNOTE;                   // [B,N,P]
    float* out_nsf  = out_phf + BATCH * NNOTE * NPART;            // [B,N]
    float* out_gate = out_nsf + BATCH * NNOTE;                    // [B,N]
    float* out_ons  = out_gate + BATCH * NNOTE;                   // [B,N]

    char* ws = (char*)d_ws;
    _Float16* A1  = (_Float16*)ws;                ws += (size_t)BATCH * NP * 2;
    _Float16* A2  = (_Float16*)ws;                ws += (size_t)BATCH * NP * 2;
    _Float16* C1t = (_Float16*)ws;                ws += (size_t)TT * NP * 2;
    _Float16* C2t = (_Float16*)ws;                ws += (size_t)TT * NP * 2;
    float*    nw  = (float*)ws;                   ws += (size_t)BATCH * NNOTE * 4;
    float*    ndp = (float*)ws;                   ws += (size_t)TT * NNOTE * 4;

    // K1: prep
    {
        int total = BATCH * NNOTE;
        piano_prep_kernel<<<(total + 255) / 256, 256, 0, stream>>>(
            note_logits, velocity, brightness, noise_amt, amp_state, phase_state,
            noise_state, prev_gate, harmonic_log, decay_logits, noise_dec_log,
            A1, A2, nw, out_ampf, out_phf, out_nsf, out_gate, out_ons);
    }
    // K2: basis
    {
        int total = NP * TT;
        piano_basis_kernel<<<(total + 255) / 256, 256, 0, stream>>>(
            decay_logits, noise_dec_log, C1t, C2t, ndp);
    }
    // K3: noise -> audio
    {
        dim3 grid((BATCH + 255) / 256, TT);
        piano_noise_kernel<<<grid, 256, 0, stream>>>(nw, ndp, noise_gain, audio);
    }
    // K4: tonal GEMM (WMMA), fused add into audio
    {
        int tiles = (BATCH / 16) * (TT / 32);   // 512 * 3 = 1536 waves
        piano_wmma_kernel<<<tiles, 32, 0, stream>>>(A1, A2, C1t, C2t, master_gain, audio);
    }
}
